// Neuralsymbol_36283883716729
// MI455X (gfx1250) — matrix operand order repaired and checked by reading, not compiled, not run on hardware
//
#include <hip/hip_runtime.h>
#include <hip/hip_bf16.h>
#include <math.h>

#define VEC 256
#define HID 1024
#define NSTACK 8
#define BB 64
#define TT 1024
#define ZOFF 1e-4f

typedef __attribute__((ext_vector_type(16))) __bf16 v16bf;
typedef __attribute__((ext_vector_type(8)))  __bf16 v8bf;
typedef __attribute__((ext_vector_type(8)))  float  v8f;

// Build a 16-element bf16 WMMA fragment from two contiguous 8-bf16 (16B) chunks.
__device__ inline v16bf frag16(const __bf16* p0, const __bf16* p1) {
    v8bf lo = *(const v8bf*)p0;
    v8bf hi = *(const v8bf*)p1;
    v16bf r;
#pragma unroll
    for (int e = 0; e < 8; ++e) { r[e] = lo[e]; r[e + 8] = hi[e]; }
    return r;
}

// ---------------- prep: transpose + convert weights to bf16 [N][K] ----------------
__global__ __launch_bounds__(256)
void ns_prep_kernel(const float* __restrict__ w1_0, const float* __restrict__ w1_1,
                    const float* __restrict__ w1_2, const float* __restrict__ w1_3,
                    const float* __restrict__ w1_4,
                    const float* __restrict__ w2b_0, const float* __restrict__ w2b_1,
                    const float* __restrict__ w2b_2,
                    const float* __restrict__ w2s_0, const float* __restrict__ w2s_1,
                    __bf16* __restrict__ W1T, __bf16* __restrict__ W2TB,
                    __bf16* __restrict__ W2TS) {
    const int N1 = 5 * HID * VEC;   // W1T:  per head [n=HID][k=VEC]
    const int N2 = 3 * VEC * HID;   // W2TB: per head [n=VEC][k=HID]
    const int N3 = 2 * 16  * HID;   // W2TS: per head [n=16 ][k=HID] (cols >=3 zero)
    for (int idx = blockIdx.x * blockDim.x + threadIdx.x; idx < N1 + N2 + N3;
         idx += gridDim.x * blockDim.x) {
        if (idx < N1) {
            int h = idx / (HID * VEC);
            int r = idx % (HID * VEC);
            int n = r / VEC, k = r % VEC;
            const float* w1 = (h == 0) ? w1_0 : (h == 1) ? w1_1 : (h == 2) ? w1_2
                                    : (h == 3) ? w1_3 : w1_4;
            W1T[idx] = (__bf16)w1[k * HID + n];
        } else if (idx < N1 + N2) {
            int i2 = idx - N1;
            int hh = i2 / (VEC * HID);
            int r = i2 % (VEC * HID);
            int n = r / HID, k = r % HID;
            const float* w2 = (hh == 0) ? w2b_0 : (hh == 1) ? w2b_1 : w2b_2;
            W2TB[i2] = (__bf16)w2[k * VEC + n];
        } else {
            int i3 = idx - N1 - N2;
            int j = i3 / (16 * HID);
            int r = i3 % (16 * HID);
            int n = r / HID, k = r % HID;
            const float* w2 = (j == 0) ? w2s_0 : w2s_1;
            W2TS[i3] = (__bf16)((n < 3) ? w2[k * 3 + n] : 0.0f);
        }
    }
}

// ---------------- init: stack state (fill + initial push of v0) -------------------
__global__ __launch_bounds__(256)
void ns_init_kernel(const float* __restrict__ op_sym,
                    float* __restrict__ cs0, float* __restrict__ cp0,
                    float* __restrict__ ws0, float* __restrict__ wp0) {
    int idx = blockIdx.x * blockDim.x + threadIdx.x;   // over 64*8*256
    if (idx >= BB * NSTACK * VEC) return;
    int v = idx % VEC;
    int n = (idx / VEC) % NSTACK;
    int b = idx / (VEC * NSTACK);
    float val = (n == 1) ? op_sym[v * 8 + 0] : ZOFF;   // after initial _push, slot1 = v0
    cs0[idx] = val;
    ws0[idx] = val;
    if (v == 0) {
        float p = (n == 1) ? 1.0f : 0.0f;
        cp0[b * NSTACK + n] = p;
        wp0[b * NSTACK + n] = p;
    }
}

// -------- fused: apply step t-1 stack update in place, soft-read, form u1/u2/u3 ---
__global__ __launch_bounds__(256)
void ns_step_kernel(float* __restrict__ cs, float* __restrict__ cp,
                    float* __restrict__ wst, float* __restrict__ wp,
                    const float* __restrict__ gc, const float* __restrict__ gw,
                    const float* __restrict__ newc, const float* __restrict__ neww,
                    const float* __restrict__ csharp, const float* __restrict__ wsharp,
                    const float* __restrict__ x, int t, int first,
                    __bf16* __restrict__ U) {
    int b = blockIdx.x;
    int v = threadIdx.x;
    float pc[NSTACK], pw[NSTACK], sc[NSTACK], sw[NSTACK];
#pragma unroll
    for (int n = 0; n < NSTACK; ++n) {
        pc[n] = cp[b * NSTACK + n];
        pw[n] = wp[b * NSTACK + n];
        sc[n] = cs[(b * NSTACK + n) * VEC + v];
        sw[n] = wst[(b * NSTACK + n) * VEC + v];
    }
    float npc[NSTACK], npw[NSTACK];
    if (!first) {
        float g0c = gc[b * 3 + 0], g1c = gc[b * 3 + 1], g2c = gc[b * 3 + 2];
        float g0w = gw[b * 3 + 0], g1w = gw[b * 3 + 1], g2w = gw[b * 3 + 2];
        float valc = newc[b * VEC + v], valw = neww[b * VEC + v];
        float shc = csharp[0], shw = wsharp[0];
        // values: s' = pop(s); s_new = g0*push(s',val) + g1*pop(s') + g2*s'
#pragma unroll
        for (int n = 0; n < NSTACK; ++n) {
            float c0 = pc[n], c1 = pc[(n + 1) & 7];
            float spop = sc[n] * (1.f - c0) + c0 * ZOFF;
            float ps = spop * (1.f - c0) + c0 * valc;          // pp == cp
            float qs = spop * (1.f - c1) + c1 * ZOFF;
            sc[n] = g0c * ps + g1c * qs + g2c * spop;
            float w0 = pw[n], w1 = pw[(n + 1) & 7];
            float wpop = sw[n] * (1.f - w0) + w0 * ZOFF;
            float pw_ = wpop * (1.f - w0) + w0 * valw;
            float qw_ = wpop * (1.f - w1) + w1 * ZOFF;
            sw[n] = g0w * pw_ + g1w * qw_ + g2w * wpop;
        }
        // pointers: pre = g0*cp + g1*roll(cp,-2) + g2*roll(cp,-1); sharpen; normalize
        float sumc = 0.f, sumw = 0.f;
#pragma unroll
        for (int n = 0; n < NSTACK; ++n) {
            float prec = g0c * pc[n] + g1c * pc[(n + 2) & 7] + g2c * pc[(n + 1) & 7];
            prec = __powf(fmaxf(prec, 0.f), shc);
            npc[n] = prec; sumc += prec;
            float prew = g0w * pw[n] + g1w * pw[(n + 2) & 7] + g2w * pw[(n + 1) & 7];
            prew = __powf(fmaxf(prew, 0.f), shw);
            npw[n] = prew; sumw += prew;
        }
        float invc = 1.f / (sumc + 1e-12f), invw = 1.f / (sumw + 1e-12f);
#pragma unroll
        for (int n = 0; n < NSTACK; ++n) { npc[n] *= invc; npw[n] *= invw; }
        __syncthreads();   // all threads finished reading cp/wp before rewrite
#pragma unroll
        for (int n = 0; n < NSTACK; ++n) {
            cs[(b * NSTACK + n) * VEC + v] = sc[n];
            wst[(b * NSTACK + n) * VEC + v] = sw[n];
        }
        if (v == 0) {
#pragma unroll
            for (int n = 0; n < NSTACK; ++n) {
                cp[b * NSTACK + n] = npc[n];
                wp[b * NSTACK + n] = npw[n];
            }
        }
    } else {
#pragma unroll
        for (int n = 0; n < NSTACK; ++n) { npc[n] = pc[n]; npw[n] = pw[n]; }
    }
    float ctrl = 0.f, wk = 0.f;
#pragma unroll
    for (int n = 0; n < NSTACK; ++n) { ctrl += sc[n] * npc[n]; wk += sw[n] * npw[n]; }
    float xt = x[((size_t)b * TT + t) * VEC + v];
    U[0 * BB * VEC + b * VEC + v] = (__bf16)(ctrl * xt);
    U[1 * BB * VEC + b * VEC + v] = (__bf16)(wk * xt);
    U[2 * BB * VEC + b * VEC + v] = (__bf16)(ctrl * wk);
}

// ---------------- B: GEMM1 (5 heads, [64x256]x[256x1024]) + bias + ReLU ----------
// 4 waves/block share the same A tile (head, mt): stage it in LDS once.
__global__ __launch_bounds__(128)
void ns_gemm1_kernel(const __bf16* __restrict__ U, const __bf16* __restrict__ W1T,
                     const float* __restrict__ b1_0, const float* __restrict__ b1_1,
                     const float* __restrict__ b1_2, const float* __restrict__ b1_3,
                     const float* __restrict__ b1_4, __bf16* __restrict__ Hbf) {
    __shared__ __attribute__((aligned(16))) __bf16 As[16 * VEC];   // 8KB
    int wave = blockIdx.x * 4 + (threadIdx.x >> 5);   // 0..1279
    int lane = threadIdx.x & 31;
    int h = wave >> 8;        // head 0..4
    int rem = wave & 255;
    int mt = rem >> 6;        // 0..3 (same for all 4 waves of a block)
    int nt = rem & 63;        // 0..63
    int ain = (h == 3) ? 1 : ((h == 4) ? 2 : 0);  // cop/ctl/wop<-u1, wrk<-u2, ff<-u3

    // cooperative A-tile stage: 16 rows x 256 cols contiguous
    {
        const v8bf* src = (const v8bf*)(U + ain * (BB * VEC) + (size_t)(mt * 16) * VEC);
        v8bf* dst = (v8bf*)As;
#pragma unroll
        for (int i = threadIdx.x; i < (16 * VEC) / 8; i += 128) dst[i] = src[i];
    }
    __syncthreads();

    int mr = lane & 15, half = lane >> 4;
    const __bf16* arow = As + mr * VEC;
    const __bf16* brow = W1T + (size_t)h * HID * VEC + (size_t)(nt * 16 + mr) * VEC;
    v8f acc = {};
#pragma unroll
    for (int ks = 0; ks < 8; ++ks) {
        int ka = ks * 32 + half * 8;     // A: K split 8/8 within 32-K step
        int kb = ks * 32 + half * 8;     // B: K split 8/8 within 32-K step
        v16bf af = frag16(arow + ka, arow + ka + 16);
        v16bf bfr = frag16(brow + kb, brow + kb + 16);
        acc = __builtin_amdgcn_wmma_f32_16x16x32_bf16(false, af, false, bfr,
                                                      (short)0, acc, false, false);
    }
    const float* b1 = (h == 0) ? b1_0 : (h == 1) ? b1_1 : (h == 2) ? b1_2
                            : (h == 3) ? b1_3 : b1_4;
    int n = lane & 15;
    int col = nt * 16 + n;
    float bias = b1[col];
    int md0 = (lane >= 16) ? 8 : 0;
#pragma unroll
    for (int r = 0; r < 8; ++r) {
        int brw = mt * 16 + md0 + r;
        float vv = acc[r] + bias;
        vv = vv > 0.f ? vv : 0.f;
        Hbf[(size_t)h * BB * HID + (size_t)brw * HID + col] = (__bf16)vv;
    }
}

// ---------------- C: GEMM2 (+bias, tanh/sigmoid), route outputs -------------------
// Big-head blocks (0..47) share one A tile per block: stage in LDS.
__global__ __launch_bounds__(128)
void ns_gemm2_kernel(const __bf16* __restrict__ Hbf, const __bf16* __restrict__ W2TB,
                     const __bf16* __restrict__ W2TS,
                     const float* __restrict__ ctl_b2, const float* __restrict__ wrk_b2,
                     const float* __restrict__ ff_b2,  const float* __restrict__ cop_b2,
                     const float* __restrict__ wop_b2,
                     float* __restrict__ newc, float* __restrict__ neww,
                     float* __restrict__ gc_ws, float* __restrict__ gw_ws,
                     float* __restrict__ out_base, int t) {
    __shared__ __attribute__((aligned(16))) __bf16 As[16 * HID];   // 32KB
    int wave = blockIdx.x * 4 + (threadIdx.x >> 5);   // 0..199
    int lane = threadIdx.x & 31;
    int mr = lane & 15, half = lane >> 4;
    int n = lane & 15;
    int md0 = (lane >= 16) ? 8 : 0;
    v8f acc = {};
    if (wave < 192) {                 // whole block uniform: blocks 0..47
        int hb = wave / 64;           // 0:ctl 1:wrk 2:ff
        int rem = wave % 64;
        int mt = rem >> 4, nt = rem & 15;    // mt shared by the block's 4 waves
        int hslot = (hb == 0) ? 1 : ((hb == 1) ? 3 : 4);
        {
            const v8bf* src =
                (const v8bf*)(Hbf + (size_t)hslot * BB * HID + (size_t)(mt * 16) * HID);
            v8bf* dst = (v8bf*)As;
#pragma unroll
            for (int i = threadIdx.x; i < (16 * HID) / 8; i += 128) dst[i] = src[i];
        }
        __syncthreads();
        const __bf16* A = As + mr * HID;
        const __bf16* Bm = W2TB + (size_t)hb * VEC * HID + (size_t)(nt * 16 + mr) * HID;
#pragma unroll 8
        for (int ks = 0; ks < 32; ++ks) {
            int ka = ks * 32 + half * 8;
            int kb = ks * 32 + half * 8;
            v16bf af = frag16(A + ka, A + ka + 16);
            v16bf bfr = frag16(Bm + kb, Bm + kb + 16);
            acc = __builtin_amdgcn_wmma_f32_16x16x32_bf16(false, af, false, bfr,
                                                          (short)0, acc, false, false);
        }
        const float* b2 = (hb == 0) ? ctl_b2 : ((hb == 1) ? wrk_b2 : ff_b2);
        int col = nt * 16 + n;
        float bias = b2[col];
#pragma unroll
        for (int r = 0; r < 8; ++r) {
            int b = mt * 16 + md0 + r;
            float vv = tanhf(acc[r] + bias);
            if (hb == 0)      newc[b * VEC + col] = vv;
            else if (hb == 1) neww[b * VEC + col] = vv;
            else              out_base[((size_t)b * TT + t) * VEC + col] = vv;  // outs
        }
    } else {                          // blocks 48..49: small padded heads, direct loads
        int wl = wave - 192;
        int sm = wl >> 2;             // 0:cop 1:wop
        int mt = wl & 3;
        int hslot = (sm == 0) ? 0 : 2;
        const __bf16* A = Hbf + (size_t)hslot * BB * HID + (size_t)(mt * 16 + mr) * HID;
        const __bf16* Bm = W2TS + (size_t)sm * 16 * HID + (size_t)mr * HID;
#pragma unroll 8
        for (int ks = 0; ks < 32; ++ks) {
            int ka = ks * 32 + half * 8;
            int kb = ks * 32 + half * 8;
            v16bf af = frag16(A + ka, A + ka + 16);
            v16bf bfr = frag16(Bm + kb, Bm + kb + 16);
            acc = __builtin_amdgcn_wmma_f32_16x16x32_bf16(false, af, false, bfr,
                                                          (short)0, acc, false, false);
        }
        if (n < 3) {
            const float* b2 = (sm == 0) ? cop_b2 : wop_b2;
            float bias = b2[n];
            float* gws = (sm == 0) ? gc_ws : gw_ws;
            size_t gbase = (size_t)BB * TT * VEC + ((sm == 1) ? (size_t)BB * TT * 3 : 0);
#pragma unroll
            for (int r = 0; r < 8; ++r) {
                int b = mt * 16 + md0 + r;
                float vv = 1.0f / (1.0f + __expf(-(acc[r] + bias)));
                gws[b * 3 + n] = vv;
                out_base[gbase + ((size_t)b * TT + t) * 3 + n] = vv;  // gc / gw
            }
        }
    }
}

extern "C" void kernel_launch(void* const* d_in, const int* in_sizes, int n_in,
                              void* d_out, int out_size, void* d_ws, size_t ws_size,
                              hipStream_t stream) {
    const float* x       = (const float*)d_in[0];
    const float* op_sym  = (const float*)d_in[1];
    const float* csharp  = (const float*)d_in[2];
    const float* wsharp  = (const float*)d_in[3];
    const float* cop_w1 = (const float*)d_in[4];  const float* cop_b1 = (const float*)d_in[5];
    const float* cop_w2 = (const float*)d_in[6];  const float* cop_b2 = (const float*)d_in[7];
    const float* ctl_w1 = (const float*)d_in[8];  const float* ctl_b1 = (const float*)d_in[9];
    const float* ctl_w2 = (const float*)d_in[10]; const float* ctl_b2 = (const float*)d_in[11];
    const float* wop_w1 = (const float*)d_in[12]; const float* wop_b1 = (const float*)d_in[13];
    const float* wop_w2 = (const float*)d_in[14]; const float* wop_b2 = (const float*)d_in[15];
    const float* wrk_w1 = (const float*)d_in[16]; const float* wrk_b1 = (const float*)d_in[17];
    const float* wrk_w2 = (const float*)d_in[18]; const float* wrk_b2 = (const float*)d_in[19];
    const float* ff_w1  = (const float*)d_in[20]; const float* ff_b1  = (const float*)d_in[21];
    const float* ff_w2  = (const float*)d_in[22]; const float* ff_b2  = (const float*)d_in[23];
    float* out = (float*)d_out;

    // ---- workspace layout (256B aligned blocks) ----
    char* ws = (char*)d_ws;
    size_t off = 0;
    auto take = [&](size_t bytes) { size_t r = off; off = (off + bytes + 255) & ~(size_t)255; return r; };
    __bf16* W1T  = (__bf16*)(ws + take((size_t)5 * HID * VEC * 2));
    __bf16* W2TB = (__bf16*)(ws + take((size_t)3 * VEC * HID * 2));
    __bf16* W2TS = (__bf16*)(ws + take((size_t)2 * 16 * HID * 2));
    __bf16* U    = (__bf16*)(ws + take((size_t)3 * BB * VEC * 2));
    __bf16* Hbf  = (__bf16*)(ws + take((size_t)5 * BB * HID * 2));
    float* newc  = (float*)(ws + take((size_t)BB * VEC * 4));
    float* neww  = (float*)(ws + take((size_t)BB * VEC * 4));
    float* gc_ws = (float*)(ws + take((size_t)BB * 3 * 4));
    float* gw_ws = (float*)(ws + take((size_t)BB * 3 * 4));
    float* CS    = (float*)(ws + take((size_t)BB * NSTACK * VEC * 4));
    float* CP    = (float*)(ws + take((size_t)BB * NSTACK * 4));
    float* WS2   = (float*)(ws + take((size_t)BB * NSTACK * VEC * 4));
    float* WP2   = (float*)(ws + take((size_t)BB * NSTACK * 4));

    // one-time weight conversion/transpose + stack init (re-run every call: deterministic)
    ns_prep_kernel<<<2080, 256, 0, stream>>>(cop_w1, ctl_w1, wop_w1, wrk_w1, ff_w1,
                                             ctl_w2, wrk_w2, ff_w2, cop_w2, wop_w2,
                                             W1T, W2TB, W2TS);
    ns_init_kernel<<<(BB * NSTACK * VEC + 255) / 256, 256, 0, stream>>>(
        op_sym, CS, CP, WS2, WP2);

    for (int t = 0; t < TT; ++t) {
        ns_step_kernel<<<BB, 256, 0, stream>>>(CS, CP, WS2, WP2,
                                               gc_ws, gw_ws, newc, neww,
                                               csharp, wsharp, x, t, (t == 0) ? 1 : 0, U);
        ns_gemm1_kernel<<<320, 128, 0, stream>>>(U, W1T, cop_b1, ctl_b1, wop_b1,
                                                 wrk_b1, ff_b1, Hbf);
        ns_gemm2_kernel<<<50, 128, 0, stream>>>(Hbf, W2TB, W2TS, ctl_b2, wrk_b2,
                                                ff_b2, cop_b2, wop_b2,
                                                newc, neww, gc_ws, gw_ws, out, t);
    }
    (void)in_sizes; (void)n_in; (void)out_size; (void)ws_size;
}
